// Block_11991548690895
// MI455X (gfx1250) — matrix operand order repaired and checked by reading, not compiled, not run on hardware
//
#include <hip/hip_runtime.h>
#include <hip/hip_bf16.h>

// ---------------------------------------------------------------------------
// Types for CDNA5 WMMA (wave32): v_wmma_f32_16x16x32_bf16
// ---------------------------------------------------------------------------
typedef __bf16 bf16;
typedef __attribute__((ext_vector_type(16))) __bf16 v16bf;
typedef __attribute__((ext_vector_type(8)))  __bf16 v8bf;
typedef __attribute__((ext_vector_type(8)))  float  v8f;
typedef __attribute__((ext_vector_type(4)))  int    v4i;

union V16 { v16bf v; v8bf h[2]; };

__device__ __forceinline__ v8f wmma_bf16(v16bf a, v16bf b, v8f c) {
  // (neg_a, A, neg_b, B, c_mod, C, reuse_a, reuse_b)
  return __builtin_amdgcn_wmma_f32_16x16x32_bf16(false, a, false, b, (short)0, c,
                                                 false, false);
}

// ---------------------------------------------------------------------------
// Async global -> LDS copy (CDNA5 GLOBAL_LOAD_ASYNC_TO_LDS_B128, ASYNCcnt),
// guarded with a fallback to plain load + ds_store.
// Signature (from hipcc diagnostic): arg0 = global v4i* (printed '__device__'),
// arg1 = LDS v4i*, then two int immediates (offset, cpol).
// ---------------------------------------------------------------------------
#define AS1 __attribute__((address_space(1)))
#define AS3 __attribute__((address_space(3)))

#if defined(__has_builtin)
# if __has_builtin(__builtin_amdgcn_global_load_async_to_lds_b128) && \
     __has_builtin(__builtin_amdgcn_s_wait_asynccnt)
#  define HAVE_ASYNC_LDS 1
# endif
#endif
#ifndef HAVE_ASYNC_LDS
# define HAVE_ASYNC_LDS 0
#endif

__device__ __forceinline__ void cp_g2lds_b128(const bf16* g, bf16* l) {
#if HAVE_ASYNC_LDS
  __builtin_amdgcn_global_load_async_to_lds_b128((AS1 v4i*)g, (AS3 v4i*)l, 0, 0);
#else
  *(v8bf*)l = *(const v8bf*)g;
#endif
}

__device__ __forceinline__ void async_wait() {
#if HAVE_ASYNC_LDS
  __builtin_amdgcn_s_wait_asynccnt(0);
#endif
}

// ---------------------------------------------------------------------------
// LayerNorm (fp32 in) -> bf16 out.  One 256-thread block (8 waves) per row.
// ---------------------------------------------------------------------------
__global__ __launch_bounds__(256)
void ln_bf16_kernel(const float* __restrict__ X, const float* __restrict__ g,
                    const float* __restrict__ bta, bf16* __restrict__ Y, int D)
{
  __shared__ float red0[8];
  __shared__ float red1[8];
  const int row = blockIdx.x;
  const float* x = X + (size_t)row * D;
  float s = 0.f, s2 = 0.f;
  for (int i = threadIdx.x; i < D; i += 256) { float v = x[i]; s += v; s2 += v * v; }
  #pragma unroll
  for (int d = 1; d < 32; d <<= 1) { s += __shfl_xor(s, d, 32); s2 += __shfl_xor(s2, d, 32); }
  const int wave = threadIdx.x >> 5, lane = threadIdx.x & 31;
  if (lane == 0) { red0[wave] = s; red1[wave] = s2; }
  __syncthreads();
  if (wave == 0) {
    float a  = (lane < 8) ? red0[lane] : 0.f;
    float b2 = (lane < 8) ? red1[lane] : 0.f;
    #pragma unroll
    for (int d = 1; d < 8; d <<= 1) { a += __shfl_xor(a, d, 32); b2 += __shfl_xor(b2, d, 32); }
    if (lane == 0) { red0[0] = a; red1[0] = b2; }
  }
  __syncthreads();
  const float mu   = red0[0] / (float)D;
  const float var  = red1[0] / (float)D - mu * mu;
  const float rstd = rsqrtf(var + 1e-5f);
  bf16* y = Y + (size_t)row * D;
  for (int i = threadIdx.x; i < D; i += 256)
    y[i] = (bf16)((x[i] - mu) * rstd * g[i] + bta[i]);
}

// ---------------------------------------------------------------------------
// Weight conversion helpers (fp32 -> bf16, TRANSPOSED), run once per launch.
// Emitting W^T makes every GEMM tile a contiguous row copy (async-friendly).
// ---------------------------------------------------------------------------
// out[n*K + k] = in[k*N + n];  K = 1<<kshift
__global__ __launch_bounds__(256)
void transpose_to_bf16_kernel(const float* __restrict__ in, bf16* __restrict__ out,
                              int kshift, int Nd, long total)
{
  long i = (long)blockIdx.x * 256 + threadIdx.x;
  if (i >= total) return;
  long n = i >> kshift;
  long k = i & ((1L << kshift) - 1);
  out[i] = (bf16)in[k * (long)Nd + n];
}

// K-slice of w_attn, transposed: wkT[n*1024 + k] = w_attn[k*3072 + 1024 + n]
__global__ __launch_bounds__(256)
void slice_wk_T_kernel(const float* __restrict__ w_attn, bf16* __restrict__ wkT)
{
  long i = (long)blockIdx.x * 256 + threadIdx.x;   // 1024*1024 total
  long n = i >> 10, k = i & 1023;
  wkT[i] = (bf16)w_attn[k * 3072 + 1024 + n];
}

// ---------------------------------------------------------------------------
// Tiled WMMA GEMM:  C(MxN, f32 acc) = A(MxK bf16, row-major) * B^T(NxK bf16)
//                   + bias[N]  (+ReLU) (+residual f32)  -> f32 and/or bf16 out
// Block tile 128x128x32, 256 threads = 8 waves in a 4(M) x 2(N) grid.
// Double-buffered LDS; tiles staged with async global->LDS copies.
// ---------------------------------------------------------------------------
template<int RELU, int RES, int WF32, int WBF>
__global__ __launch_bounds__(256)
void gemm_bf16_kernel(const bf16* __restrict__ A, const bf16* __restrict__ BT,
                      const float* __restrict__ bias, const float* __restrict__ res,
                      float* __restrict__ Cf, bf16* __restrict__ Cb,
                      int M, int N, int K)
{
  constexpr int BM = 128, BN = 128, BK = 32, ST = BK + 8;  // 80-byte rows (16B aligned)
  __shared__ bf16 As[2][BM][ST];   // [buf][m][k]
  __shared__ bf16 Bs[2][BN][ST];   // [buf][n][k]

  const int t    = threadIdx.x;
  const int wave = t >> 5, lane = t & 31;
  const int half = lane >> 4, r = lane & 15;
  const int wm   = wave >> 1, wn = wave & 1;
  const int m0   = blockIdx.y * BM, n0 = blockIdx.x * BN;

  v8f acc[2][4] = {};

  // stage one 128x32 tile pair into LDS buffer `buf`
  auto stage = [&](int k0, int buf) {
    #pragma unroll
    for (int i = 0; i < 2; i++) {
      int c = t + i * 256;
      int row = c >> 2, col = (c & 3) * 8;
      cp_g2lds_b128(A  + (size_t)(m0 + row) * K + k0 + col, &As[buf][row][col]);
      cp_g2lds_b128(BT + (size_t)(n0 + row) * K + k0 + col, &Bs[buf][row][col]);
    }
  };

  stage(0, 0);
  async_wait();
  __syncthreads();

  for (int k0 = 0; k0 < K; k0 += BK) {
    const int cur = (k0 >> 5) & 1;
    // prefetch tile k+2 into the cache hierarchy, async-stage tile k+1
    if (k0 + 2 * BK < K) {
      __builtin_prefetch(A  + (size_t)(m0 + (t >> 1)) * K + k0 + 2 * BK, 0, 0);
      __builtin_prefetch(BT + (size_t)(n0 + (t >> 1)) * K + k0 + 2 * BK, 0, 0);
    }
    if (k0 + BK < K) stage(k0 + BK, cur ^ 1);

    // ---- fragments (ISA 16-bit A layout: half 0 -> K 0-7|16-23, half 1 -> K 8-15|24-31;
    //      B layout: lanes = columns, same K order as A) ----
    V16 af[2];
    #pragma unroll
    for (int mt = 0; mt < 2; mt++) {
      const bf16* p = &As[cur][wm * 32 + mt * 16 + r][half * 8];
      af[mt].h[0] = *(const v8bf*)p;
      af[mt].h[1] = *(const v8bf*)(p + 16);
    }
    #pragma unroll
    for (int nt = 0; nt < 4; nt++) {
      V16 bfr;
      const bf16* p = &Bs[cur][wn * 64 + nt * 16 + r][half * 8];
      bfr.h[0] = *(const v8bf*)p;
      bfr.h[1] = *(const v8bf*)(p + 16);
      #pragma unroll
      for (int mt = 0; mt < 2; mt++)
        acc[mt][nt] = wmma_bf16(af[mt].v, bfr.v, acc[mt][nt]);
    }

    async_wait();       // next tile fully resident in LDS
    __syncthreads();    // all waves done with `cur`, next tile visible
  }

  // ---- epilogue: bias (+relu) (+residual) -> f32 / bf16 ----
  #pragma unroll
  for (int mt = 0; mt < 2; mt++) {
    #pragma unroll
    for (int nt = 0; nt < 4; nt++) {
      const int gn = n0 + wn * 64 + nt * 16 + r;
      const float bv = bias[gn];
      #pragma unroll
      for (int i = 0; i < 8; i++) {
        const int gm = m0 + wm * 32 + mt * 16 + half * 8 + i;
        float v = acc[mt][nt][i] + bv;
        if (RELU) v = v > 0.f ? v : 0.f;
        if (RES)  v += res[(size_t)gm * N + gn];
        if (WF32) Cf[(size_t)gm * N + gn] = v;
        if (WBF)  Cb[(size_t)gm * N + gn] = (bf16)v;
      }
    }
  }
}

// ---------------------------------------------------------------------------
// Flash attention with q = k = v (reference quirk).  dh = 64, scale = 1/8.
// Grid: (L/64, H, B); 128 threads = 4 waves, each wave owns 16 query rows.
// Streams 64-key tiles; S and P*V via WMMA; online softmax in f32.
// ---------------------------------------------------------------------------
__global__ __launch_bounds__(128)
void attn_kernel(const bf16* __restrict__ Km, bf16* __restrict__ Out, int L)
{
  constexpr int DH = 64, TQ = 64, TK = 64, ST = 72;  // 144-byte rows (16B aligned)
  __shared__ bf16 Qs [TQ][ST];     // [q][d]
  __shared__ bf16 Ks [TK][ST];     // [key][d]  (B-frag for S: lanes = key cols)
  __shared__ bf16 KsT[DH][ST];     // [d][key]  (B-frag for P*V: lanes = d cols)
  __shared__ bf16 Ps [4][16][ST];  // per-wave P tile, [q][key] (A-frag layout source)

  const int t    = threadIdx.x;
  const int wave = t >> 5, lane = t & 31;
  const int half = lane >> 4, r = lane & 15;
  const int q0 = blockIdx.x * TQ;
  const int h  = blockIdx.y;
  const int b  = blockIdx.z;
  const bf16* base = Km + ((size_t)b * L) * 1024 + h * DH;  // row stride 1024

  // load Q tile (64x64)
  #pragma unroll
  for (int i = 0; i < 4; i++) {
    int c = t + i * 128;
    int row = c >> 3, col = (c & 7) * 8;
    *(v8bf*)&Qs[row][col] = *(const v8bf*)(base + (size_t)(q0 + row) * 1024 + col);
  }

  v8f  o[4] = {};
  float mi[8], li[8];
  #pragma unroll
  for (int i = 0; i < 8; i++) { mi[i] = -3.0e38f; li[i] = 0.f; }

  for (int kt = 0; kt < L; kt += TK) {
    __syncthreads();
    // stage K/V tile (natural + transposed)
    #pragma unroll
    for (int i = 0; i < 4; i++) {
      int c = t + i * 128;
      int row = c >> 3, col = (c & 7) * 8;
      v8bf kv = *(const v8bf*)(base + (size_t)(kt + row) * 1024 + col);
      *(v8bf*)&Ks[row][col] = kv;
      #pragma unroll
      for (int j = 0; j < 8; j++) KsT[col + j][row] = kv[j];
    }
    __syncthreads();

    // S = Q * K^T  (16x64 per wave; dh contraction in 2 steps of 32)
    v8f s[4] = {};
    #pragma unroll
    for (int ks = 0; ks < 2; ks++) {
      V16 aq;
      const bf16* p = &Qs[wave * 16 + r][ks * 32 + half * 8];
      aq.h[0] = *(const v8bf*)p;  aq.h[1] = *(const v8bf*)(p + 16);
      #pragma unroll
      for (int nt = 0; nt < 4; nt++) {
        V16 kb;
        const bf16* pb = &Ks[nt * 16 + r][ks * 32 + half * 8];
        kb.h[0] = *(const v8bf*)pb;  kb.h[1] = *(const v8bf*)(pb + 16);
        s[nt] = wmma_bf16(aq.v, kb.v, s[nt]);
      }
    }

    // online softmax (C-layout: row = half*8+i, cols across 16 lanes of the half)
    #pragma unroll
    for (int i = 0; i < 8; i++) {
      float mx = -3.0e38f;
      #pragma unroll
      for (int nt = 0; nt < 4; nt++) mx = fmaxf(mx, s[nt][i] * 0.125f);
      #pragma unroll
      for (int d = 1; d < 16; d <<= 1) mx = fmaxf(mx, __shfl_xor(mx, d, 32));
      const float newm  = fmaxf(mi[i], mx);
      const float alpha = __expf(mi[i] - newm);
      float rs = 0.f;
      #pragma unroll
      for (int nt = 0; nt < 4; nt++) {
        float v = __expf(s[nt][i] * 0.125f - newm);
        s[nt][i] = v;
        rs += v;
      }
      #pragma unroll
      for (int d = 1; d < 16; d <<= 1) rs += __shfl_xor(rs, d, 32);
      li[i] = li[i] * alpha + rs;
      mi[i] = newm;
      #pragma unroll
      for (int nt = 0; nt < 4; nt++) o[nt][i] *= alpha;
    }

    // round-trip P through per-wave LDS to reach the A-fragment layout
    #pragma unroll
    for (int nt = 0; nt < 4; nt++)
      #pragma unroll
      for (int i = 0; i < 8; i++)
        Ps[wave][half * 8 + i][nt * 16 + r] = (bf16)s[nt][i];

    // O += P * V   (key contraction in 2 steps of 32; V = K tile, transposed copy)
    #pragma unroll
    for (int ks = 0; ks < 2; ks++) {
      V16 ap;
      const bf16* p = &Ps[wave][r][ks * 32 + half * 8];
      ap.h[0] = *(const v8bf*)p;  ap.h[1] = *(const v8bf*)(p + 16);
      #pragma unroll
      for (int nt = 0; nt < 4; nt++) {
        V16 bv;
        const bf16* pb = &KsT[nt * 16 + r][ks * 32 + half * 8];
        bv.h[0] = *(const v8bf*)pb;  bv.h[1] = *(const v8bf*)(pb + 16);
        o[nt] = wmma_bf16(ap.v, bv.v, o[nt]);
      }
    }
  }

  // normalize and store (layout matches GEMM consumer: [b*L + q][h*64 + d])
  bf16* obase = Out + ((size_t)b * L + q0 + wave * 16) * 1024 + h * DH;
  #pragma unroll
  for (int i = 0; i < 8; i++) {
    const float inv = 1.f / li[i];
    #pragma unroll
    for (int nt = 0; nt < 4; nt++)
      obase[(size_t)(half * 8 + i) * 1024 + nt * 16 + r] = (bf16)(o[nt][i] * inv);
  }
}

// ---------------------------------------------------------------------------
// Host-side orchestration
// ---------------------------------------------------------------------------
extern "C" void kernel_launch(void* const* d_in, const int* in_sizes, int n_in,
                              void* d_out, int out_size, void* d_ws, size_t ws_size,
                              hipStream_t stream)
{
  (void)in_sizes; (void)n_in; (void)out_size; (void)ws_size;
  const float* x      = (const float*)d_in[0];
  const float* ln1_g  = (const float*)d_in[1];
  const float* ln1_b  = (const float*)d_in[2];
  const float* w_attn = (const float*)d_in[3];
  const float* b_attn = (const float*)d_in[4];
  const float* w_proj = (const float*)d_in[5];
  const float* b_proj = (const float*)d_in[6];
  const float* ln2_g  = (const float*)d_in[7];
  const float* ln2_b  = (const float*)d_in[8];
  const float* w_fc1  = (const float*)d_in[9];
  const float* b_fc1  = (const float*)d_in[10];
  const float* w_fc2  = (const float*)d_in[11];
  const float* b_fc2  = (const float*)d_in[12];

  const int B = 2, L = 2048, D = 1024, H = 16, M = B * L, F = 4 * D;

  char* ws = (char*)d_ws;
  size_t off = 0;
  auto alloc = [&](size_t bytes) -> void* {
    void* p = ws + off;
    off += (bytes + 255) & ~(size_t)255;
    return p;
  };
  bf16*  h1   = (bf16*)alloc((size_t)M * D * 2);   // LN1 output
  bf16*  wkT  = (bf16*)alloc((size_t)D * D * 2);   // (K slice of w_attn)^T   [n][k]
  bf16*  kbf  = (bf16*)alloc((size_t)M * D * 2);   // K projection (q=k=v)
  bf16*  outb = (bf16*)alloc((size_t)M * D * 2);   // attention output
  bf16*  wpT  = (bf16*)alloc((size_t)D * D * 2);   // w_proj^T
  float* x1   = (float*)alloc((size_t)M * D * 4);  // x + attn proj (residual 1)
  bf16*  h2   = (bf16*)alloc((size_t)M * D * 2);   // LN2 output
  bf16*  wf1T = (bf16*)alloc((size_t)D * F * 2);   // w_fc1^T  [F][D]
  bf16*  fbuf = (bf16*)alloc((size_t)M * F * 2);   // relu(fc1)
  bf16*  wf2T = (bf16*)alloc((size_t)F * D * 2);   // w_fc2^T  [D][F]

  // --- weight conversions (transposed so GEMM tiles are contiguous) ---
  slice_wk_T_kernel<<<(D * D) / 256, 256, 0, stream>>>(w_attn, wkT);
  transpose_to_bf16_kernel<<<((long)D * D + 255) / 256, 256, 0, stream>>>(
      w_proj, wpT, 10, D, (long)D * D);
  transpose_to_bf16_kernel<<<((long)D * F + 255) / 256, 256, 0, stream>>>(
      w_fc1, wf1T, 10, F, (long)D * F);
  transpose_to_bf16_kernel<<<((long)F * D + 255) / 256, 256, 0, stream>>>(
      w_fc2, wf2T, 12, D, (long)F * D);

  // --- LN1 ---
  ln_bf16_kernel<<<M, 256, 0, stream>>>(x, ln1_g, ln1_b, h1, D);

  // --- K projection: kbf = h1 @ wk + b_attn[D:2D]  (bf16 out) ---
  gemm_bf16_kernel<0, 0, 0, 1><<<dim3(D / 128, M / 128), 256, 0, stream>>>(
      h1, wkT, b_attn + D, nullptr, nullptr, kbf, M, D, D);

  // --- attention (q = k = v = kbf) ---
  attn_kernel<<<dim3(L / 64, H, B), 128, 0, stream>>>(kbf, outb, L);

  // --- x1 = x + outb @ w_proj + b_proj  (f32 out) ---
  gemm_bf16_kernel<0, 1, 1, 0><<<dim3(D / 128, M / 128), 256, 0, stream>>>(
      outb, wpT, b_proj, x, x1, nullptr, M, D, D);

  // --- LN2 ---
  ln_bf16_kernel<<<M, 256, 0, stream>>>(x1, ln2_g, ln2_b, h2, D);

  // --- fbuf = relu(h2 @ w_fc1 + b_fc1)  (bf16 out) ---
  gemm_bf16_kernel<1, 0, 0, 1><<<dim3(F / 128, M / 128), 256, 0, stream>>>(
      h2, wf1T, b_fc1, nullptr, nullptr, fbuf, M, F, D);

  // --- out = x1 + fbuf @ w_fc2 + b_fc2  (f32 out to d_out) ---
  gemm_bf16_kernel<0, 1, 1, 0><<<dim3(D / 128, M / 128), 256, 0, stream>>>(
      fbuf, wf2T, b_fc2, x1, (float*)d_out, nullptr, M, D, F);
}
